// SimpleHeteroGNN_52802327937193
// MI455X (gfx1250) — matrix operand order repaired and checked by reading, not compiled, not run on hardware
//
#include <hip/hip_runtime.h>
#include <hip/hip_bf16.h>

typedef __attribute__((ext_vector_type(16))) _Float16 v16h;
typedef __attribute__((ext_vector_type(8)))  _Float16 h8;
typedef __attribute__((ext_vector_type(4)))  _Float16 h4;
typedef __attribute__((ext_vector_type(8)))  float    v8f;

#define NVN   100000
#define NEN   100000
#define NEDGE 250000
#define HID   128
#define HEADS 4
#define NSRC  512   // HEADS * DH

// ---- order-preserving float <-> uint encoding for atomic max ----
__device__ __forceinline__ unsigned fenc(float f) {
  unsigned u = __float_as_uint(f);
  return (u & 0x80000000u) ? ~u : (u | 0x80000000u);
}
__device__ __forceinline__ float fdec(unsigned k) {
  unsigned u = (k & 0x80000000u) ? (k & 0x7fffffffu) : ~k;
  return __uint_as_float(u);
}

// ---- convert + transpose weights: Wt[n*K + k] = (f16) W[k*N + n] ----
__global__ void k_cvt_t(const float* __restrict__ W, _Float16* __restrict__ Wt,
                        int K, int N) {
  int idx = blockIdx.x * 256 + threadIdx.x;
  if (idx >= K * N) return;
  int n = idx / K, k = idx - n * K;
  Wt[idx] = (_Float16)W[(size_t)k * N + n];
}

// ---- init per-call scratch: acc=0, denom=0, smax keys = enc(-3e38) ----
__global__ void k_init(float* __restrict__ acc, float* __restrict__ denom,
                       unsigned* __restrict__ smaxkey) {
  int i = blockIdx.x * 256 + threadIdx.x;
  if (i < NEN * HID) acc[i] = 0.f;
  if (i < NEN * HEADS) { denom[i] = 0.f; smaxkey[i] = fenc(-3.0e38f); }
}

// ---- node encoder: h = LayerNorm(ReLU(x @ W + b)) -> f16, wave per node ----
__global__ __launch_bounds__(256)
void k_encode(const float* __restrict__ x, const float* __restrict__ W,
              const float* __restrict__ b, const float* __restrict__ g,
              const float* __restrict__ beta, _Float16* __restrict__ h16, int M) {
  __shared__ float sW[16 * HID];
  int tid = threadIdx.x;
  for (int i = tid; i < 16 * HID; i += 256) sW[i] = W[i];
  __syncthreads();
  int wave = tid >> 5, lane = tid & 31;
  int node = blockIdx.x * 8 + wave;
  if (node >= M) return;
  const float* xr = x + (size_t)node * 16;
  float xv[16];
#pragma unroll
  for (int k = 0; k < 16; ++k) xv[k] = xr[k];
  float4 a = ((const float4*)b)[lane];
#pragma unroll
  for (int k = 0; k < 16; ++k) {
    float4 w = ((const float4*)(sW + k * HID))[lane];
    a.x += xv[k] * w.x; a.y += xv[k] * w.y;
    a.z += xv[k] * w.z; a.w += xv[k] * w.w;
  }
  a.x = fmaxf(a.x, 0.f); a.y = fmaxf(a.y, 0.f);
  a.z = fmaxf(a.z, 0.f); a.w = fmaxf(a.w, 0.f);
  float s = a.x + a.y + a.z + a.w;
#pragma unroll
  for (int off = 16; off; off >>= 1) s += __shfl_xor(s, off, 32);
  float mu = s * (1.0f / HID);
  float dx = a.x - mu, dy = a.y - mu, dz = a.z - mu, dw = a.w - mu;
  float var = dx * dx + dy * dy + dz * dz + dw * dw;
#pragma unroll
  for (int off = 16; off; off >>= 1) var += __shfl_xor(var, off, 32);
  float rs = rsqrtf(var * (1.0f / HID) + 1e-5f);
  float4 gg = ((const float4*)g)[lane];
  float4 bb = ((const float4*)beta)[lane];
  h4 o;
  o[0] = (_Float16)(dx * rs * gg.x + bb.x);
  o[1] = (_Float16)(dy * rs * gg.y + bb.y);
  o[2] = (_Float16)(dz * rs * gg.z + bb.z);
  o[3] = (_Float16)(dw * rs * gg.w + bb.w);
  *(h4*)(h16 + (size_t)node * HID + lane * 4) = o;
}

// ---- WMMA GEMM: C[M,N] = act(A[M,128] @ B[128,N] + bias), K fixed = 128 ----
// A: f16 row-major (or f32 converted inline).  Bt: f16, Bt[n*128+k] = B[k][n].
// Block: 256 threads = 8 waves; wave tile 16M x 64N; block tile 64M x 128N.
template<bool AF32, bool RELU, bool OUTF16>
__global__ __launch_bounds__(256)
void k_gemm(const void* __restrict__ Ap, const _Float16* __restrict__ Bt,
            const float* __restrict__ bias, void* __restrict__ Cp,
            int M, int N) {
  int tid = threadIdx.x;
  int wave = tid >> 5, lane = tid & 31;
  int wm = wave >> 1, wn = wave & 1;
  int m0 = blockIdx.y * 64 + wm * 16;
  int n0 = blockIdx.x * 128 + wn * 64;
  if (m0 >= M) return;                 // M % 16 == 0: tiles fully valid
  int  lrow = lane & 15;
  bool hiH  = lane >= 16;

  v8f c[4] = {v8f{}, v8f{}, v8f{}, v8f{}};
#pragma unroll
  for (int kb = 0; kb < 128; kb += 32) {
    // A fragment (16x32 f16 layout): lane<16 -> K kb+0..7 / kb+16..23,
    //                                lane>=16 -> K kb+8..15 / kb+24..31
    int klo = kb + (hiH ? 8 : 0);
    v16h a;
    if (AF32) {
      const float* Ar = (const float*)Ap + (size_t)(m0 + lrow) * 128;
#pragma unroll
      for (int i = 0; i < 8; ++i) a[i]     = (_Float16)Ar[klo + i];
#pragma unroll
      for (int i = 0; i < 8; ++i) a[8 + i] = (_Float16)Ar[klo + 16 + i];
    } else {
      const _Float16* Ar = (const _Float16*)Ap + (size_t)(m0 + lrow) * 128;
      h8 lo = *(const h8*)(Ar + klo);
      h8 hh = *(const h8*)(Ar + klo + 16);
#pragma unroll
      for (int i = 0; i < 8; ++i) { a[i] = lo[i]; a[8 + i] = hh[i]; }
    }
    // B fragments (32x16): lanes 0-15 hold K kb+0..7 / kb+16..23, lanes 16-31 K kb+8..15 / kb+24..31,
    // column = n0 + 16j + (lane&15); Bt rows are contiguous in K.
    int kbb = kb + (hiH ? 8 : 0);
    const _Float16* B0 = Bt + (size_t)(n0 + lrow) * 128 + kbb;
#pragma unroll
    for (int j = 0; j < 4; ++j) {
      h8 blo = *(const h8*)(B0 + (size_t)j * 16 * 128);
      h8 bhi = *(const h8*)(B0 + (size_t)j * 16 * 128 + 16);
      v16h bfr;
#pragma unroll
      for (int i = 0; i < 8; ++i) { bfr[i] = blo[i]; bfr[8 + i] = bhi[i]; }
      c[j] = __builtin_amdgcn_wmma_f32_16x16x32_f16(
          false, a, false, bfr, (short)0, c[j], false, false);
    }
  }
  // C/D layout: VGPR i -> row m0+i (lanes 0-15) / m0+8+i (lanes 16-31), col=lane&15
  int rbase = m0 + (hiH ? 8 : 0);
#pragma unroll
  for (int j = 0; j < 4; ++j) {
    int col = n0 + 16 * j + lrow;
    float bs = bias[col];
#pragma unroll
    for (int i = 0; i < 8; ++i) {
      float v = c[j][i] + bs;
      if (RELU) v = fmaxf(v, 0.f);
      size_t idx = (size_t)(rbase + i) * N + col;
      if (OUTF16) ((_Float16*)Cp)[idx] = (_Float16)v;
      else        ((float*)Cp)[idx]    = v;
    }
  }
}

// ---- edge scores + segment max: wave per edge ----
__global__ __launch_bounds__(256)
void k_score(const _Float16* __restrict__ fs, const _Float16* __restrict__ fd,
             const float* __restrict__ attn, const int* __restrict__ src,
             const int* __restrict__ dst, float* __restrict__ score,
             unsigned* __restrict__ smaxkey) {
  int wave = threadIdx.x >> 5, lane = threadIdx.x & 31;
  int e = blockIdx.x * 8 + wave;
  if (e >= NEDGE) return;
  int s = src[e], d = dst[e];
  const _Float16* fsr = fs + (size_t)s * NSRC;
  const _Float16* fdr = fd + (size_t)d * NSRC;
#pragma unroll
  for (int h = 0; h < HEADS; ++h) {
    int o = h * HID + lane * 4;
    h4 fa = *(const h4*)(fsr + o);
    h4 fb = *(const h4*)(fdr + o);
    float4 at = *(const float4*)(attn + o);
    float aw[4] = {at.x, at.y, at.z, at.w};
    float p = 0.f;
#pragma unroll
    for (int i = 0; i < 4; ++i) {
      float v = (float)fa[i] + (float)fb[i];
      v = v > 0.f ? v : 0.2f * v;         // leaky relu
      p += aw[i] * v;
    }
#pragma unroll
    for (int off = 16; off; off >>= 1) p += __shfl_xor(p, off, 32);
    if (lane == 0) {
      score[(size_t)e * HEADS + h] = p;
      atomicMax(smaxkey + (size_t)d * HEADS + h, fenc(p));
    }
  }
}

// ---- exp + segment sum: thread per (edge, head) ----
__global__ void k_exp(const float* __restrict__ score, const int* __restrict__ dst,
                      const unsigned* __restrict__ smaxkey,
                      float* __restrict__ ex, float* __restrict__ denom) {
  int idx = blockIdx.x * 256 + threadIdx.x;
  if (idx >= NEDGE * HEADS) return;
  int e = idx >> 2, h = idx & 3;
  int d = dst[e];
  float ev = expf(score[idx] - fdec(smaxkey[(size_t)d * HEADS + h]));
  ex[idx] = ev;
  atomicAdd(denom + (size_t)d * HEADS + h, ev);
}

// ---- weighted aggregation fused with head-mean: wave per edge ----
__global__ __launch_bounds__(256)
void k_agg(const _Float16* __restrict__ fs, const float* __restrict__ ex,
           const float* __restrict__ denom, const int* __restrict__ src,
           const int* __restrict__ dst, float* __restrict__ acc) {
  int wave = threadIdx.x >> 5, lane = threadIdx.x & 31;
  int e = blockIdx.x * 8 + wave;
  if (e >= NEDGE) return;
  int s = src[e], d = dst[e];
  float alpha[HEADS];
#pragma unroll
  for (int h = 0; h < HEADS; ++h)
    alpha[h] = ex[(size_t)e * HEADS + h] / denom[(size_t)d * HEADS + h];
  const _Float16* fsr = fs + (size_t)s * NSRC;
  float v[4] = {0.f, 0.f, 0.f, 0.f};
#pragma unroll
  for (int h = 0; h < HEADS; ++h) {
    h4 fa = *(const h4*)(fsr + h * HID + lane * 4);
#pragma unroll
    for (int i = 0; i < 4; ++i) v[i] += alpha[h] * (float)fa[i];
  }
  float* ar = acc + (size_t)d * HID + lane * 4;
#pragma unroll
  for (int i = 0; i < 4; ++i) atomicAdd(ar + i, 0.25f * v[i]);
}

extern "C" void kernel_launch(void* const* d_in, const int* in_sizes, int n_in,
                              void* d_out, int out_size, void* d_ws, size_t ws_size,
                              hipStream_t stream) {
  const float* v_feat = (const float*)d_in[0];
  const float* e_feat = (const float*)d_in[1];
  const float* Wv     = (const float*)d_in[2];
  const float* bv     = (const float*)d_in[3];
  const float* gv     = (const float*)d_in[4];
  const float* betav  = (const float*)d_in[5];
  const float* We     = (const float*)d_in[6];
  const float* be     = (const float*)d_in[7];
  const float* ge     = (const float*)d_in[8];
  const float* betae  = (const float*)d_in[9];
  const float* Wsrc   = (const float*)d_in[10];
  const float* bsrc   = (const float*)d_in[11];
  const float* Wdst   = (const float*)d_in[12];
  const float* bdst   = (const float*)d_in[13];
  const float* attn   = (const float*)d_in[14];
  const float* Wp     = (const float*)d_in[15];
  const float* bp     = (const float*)d_in[16];
  const int*   src    = (const int*)d_in[17];
  const int*   dst    = (const int*)d_in[18];
  (void)in_sizes; (void)n_in; (void)out_size; (void)ws_size;

  char* ws = (char*)d_ws;
  size_t off = 0;
  auto take = [&](size_t bytes) -> char* {
    char* p = ws + off;
    off = (off + bytes + 255) & ~(size_t)255;
    return p;
  };
  _Float16* fs16    = (_Float16*)take((size_t)NVN * NSRC * 2);
  _Float16* fd16    = (_Float16*)take((size_t)NEN * NSRC * 2);
  _Float16* hv16    = (_Float16*)take((size_t)NVN * HID * 2);
  _Float16* he16    = (_Float16*)take((size_t)NEN * HID * 2);
  _Float16* WsrcT   = (_Float16*)take((size_t)HID * NSRC * 2);
  _Float16* WdstT   = (_Float16*)take((size_t)HID * NSRC * 2);
  _Float16* WpT     = (_Float16*)take((size_t)HID * HID * 2);
  float*    score   = (float*)take((size_t)NEDGE * HEADS * 4);
  float*    exb     = (float*)take((size_t)NEDGE * HEADS * 4);
  unsigned* smaxkey = (unsigned*)take((size_t)NEN * HEADS * 4);
  float*    denom   = (float*)take((size_t)NEN * HEADS * 4);
  float*    acc     = (float*)take((size_t)NEN * HID * 4);

  // weight convert/transpose (tiny)
  k_cvt_t<<<(HID * NSRC + 255) / 256, 256, 0, stream>>>(Wsrc, WsrcT, HID, NSRC);
  k_cvt_t<<<(HID * NSRC + 255) / 256, 256, 0, stream>>>(Wdst, WdstT, HID, NSRC);
  k_cvt_t<<<(HID * HID + 255) / 256, 256, 0, stream>>>(Wp, WpT, HID, HID);

  // per-call scratch init
  k_init<<<(NEN * HID + 255) / 256, 256, 0, stream>>>(acc, denom, smaxkey);

  // node encoders
  k_encode<<<(NVN + 7) / 8, 256, 0, stream>>>(v_feat, Wv, bv, gv, betav, hv16, NVN);
  k_encode<<<(NEN + 7) / 8, 256, 0, stream>>>(e_feat, We, be, ge, betae, he16, NEN);

  // fs / fd projections via WMMA
  dim3 g1(NSRC / 128, (NVN + 63) / 64);
  k_gemm<false, false, true><<<g1, 256, 0, stream>>>(hv16, WsrcT, bsrc, fs16, NVN, NSRC);
  dim3 g2(NSRC / 128, (NEN + 63) / 64);
  k_gemm<false, false, true><<<g2, 256, 0, stream>>>(he16, WdstT, bdst, fd16, NEN, NSRC);

  // GATv2 segment softmax + aggregation
  k_score<<<(NEDGE + 7) / 8, 256, 0, stream>>>(fs16, fd16, attn, src, dst, score, smaxkey);
  k_exp<<<(NEDGE * HEADS + 255) / 256, 256, 0, stream>>>(score, dst, smaxkey, exb, denom);
  k_agg<<<(NEDGE + 7) / 8, 256, 0, stream>>>(fs16, exb, denom, src, dst, acc);

  // final projection + ReLU via WMMA (A converted f32->f16 inline)
  dim3 g3(HID / 128, (NEN + 63) / 64);
  k_gemm<true, true, false><<<g3, 256, 0, stream>>>(acc, WpT, bp, d_out, NEN, HID);
}
